// ManualRNN_60138132079089
// MI455X (gfx1250) — matrix operand order repaired and checked by reading, not compiled, not run on hardware
//
#include <hip/hip_runtime.h>
#include <hip/hip_bf16.h>

typedef __bf16 bf16_t;
typedef __attribute__((ext_vector_type(16))) __bf16 v16bf;
typedef __attribute__((ext_vector_type(8)))  __bf16 v8bf;
typedef __attribute__((ext_vector_type(8)))  float  v8f;

#define T_STEPS   2048
#define BATCH     64
#define I_DIM     256
#define H_DIM     512
#define K_AUG     768            // H_DIM + I_DIM (augmented K)
#define KP        776            // padded LDS K-stride (bf16 elems); 776*2 = 97*16 bytes
#define COLS_WG   128            // H columns owned per workgroup
#define ROW_TILE  16             // batch rows per row-group
#define WG_THREADS 256           // 8 wave32
#define KCHUNKS   (K_AUG / 32)   // 24 WMMA k-steps

__device__ __forceinline__ float fast_tanh(float x) {
#if __has_builtin(__builtin_amdgcn_tanhf)
  return __builtin_amdgcn_tanhf(x);
#elif __has_builtin(__builtin_amdgcn_tanh_f32)
  return __builtin_amdgcn_tanh_f32(x);
#else
  return tanhf(x);
#endif
}

__global__ void rnn_init_counters(unsigned* counters) {
  if (threadIdx.x < 4) counters[threadIdx.x] = 0u;
}

// Grid: 16 WGs = 4 row-groups (independent) x 4 column-groups (barriered).
// Each WG: rows [16*rg, 16*rg+16), cols [128*cg, 128*cg+128).
// LDS: Wl[128][KP] bf16 (resident augmented weights), Al[16][KP] bf16 ([h_t | x_t]).
__global__ __launch_bounds__(WG_THREADS) void rnn_scan(
    const float* __restrict__ input,   // [T, B, I]
    const float* __restrict__ w_in,    // [I, H]
    const float* __restrict__ w_rec,   // [H, H]
    float*       __restrict__ out,     // [T*B*H] hiddens ++ [B*H] h_last
    unsigned*    __restrict__ counters,// [4] one per row-group
    bf16_t*      __restrict__ hbuf)    // [4 rowgroups][2 parity][16][512] bf16
{
  extern __shared__ __align__(16) char smem[];
  bf16_t* Wl = (bf16_t*)smem;                                  // [COLS_WG][KP]
  bf16_t* Al = (bf16_t*)(smem + (size_t)COLS_WG * KP * 2);     // [ROW_TILE][KP]

  const int tid     = threadIdx.x;
  const int rg      = blockIdx.x >> 2;       // row group 0..3
  const int cg      = blockIdx.x & 3;        // col group 0..3
  const int rowbase = rg * ROW_TILE;
  const int colbase = cg * COLS_WG;
  const int lane    = tid & 31;
  const int wv      = tid >> 5;              // wave 0..7 -> N-tile
  const int n       = lane & 15;             // M row (A) / N col (B,D)
  const int hi      = lane >> 4;             // half-wave select

  // ---- one-time: stage augmented weight slice, column-major, f32 -> bf16 ----
  // Wl[col][k]: k<512 -> w_rec[k][gcol], k>=512 -> w_in[k-512][gcol]
  for (int idx = tid; idx < COLS_WG * K_AUG; idx += WG_THREADS) {
    int col  = idx & (COLS_WG - 1);
    int k    = idx >> 7;
    int gcol = colbase + col;
    float w  = (k < H_DIM) ? w_rec[(size_t)k * H_DIM + gcol]
                           : w_in[(size_t)(k - H_DIM) * H_DIM + gcol];
    Wl[col * KP + k] = (bf16_t)w;
  }

  // ---- h_0 = 0: zero our column slice of parity-0 h buffer ----
  bf16_t* hb0 = hbuf + ((size_t)rg * 2 + 0) * (ROW_TILE * H_DIM);
  for (int idx = tid; idx < ROW_TILE * COLS_WG; idx += WG_THREADS) {
    int m = idx >> 7;
    int c = idx & (COLS_WG - 1);
    hb0[m * H_DIM + colbase + c] = (bf16_t)0.0f;
  }
  __threadfence();
  __syncthreads();
  if (tid == 0)
    __hip_atomic_fetch_add(counters + rg, 1u, __ATOMIC_RELEASE, __HIP_MEMORY_SCOPE_AGENT);

  for (int t = 0; t < T_STEPS; ++t) {
    // ---- wait until all 4 column-WGs of this row-group published h_t ----
    if (tid == 0) {
      const unsigned target = 4u * (unsigned)(t + 1);
      while (__hip_atomic_load(counters + rg, __ATOMIC_RELAXED,
                               __HIP_MEMORY_SCOPE_AGENT) < target)
        __builtin_amdgcn_s_sleep(1);
    }
    __syncthreads();
    __threadfence();   // acquire side: drop stale hbuf lines in near caches

    // ---- stage A = [h_t | x_t] into LDS (bf16, row-major, stride KP) ----
    const bf16_t*   hin   = hbuf + ((size_t)rg * 2 + (t & 1)) * (ROW_TILE * H_DIM);
    const unsigned* hin32 = (const unsigned*)hin;
    for (int idx = tid; idx < ROW_TILE * (H_DIM / 2); idx += WG_THREADS) {
      int m  = idx >> 8;               // 256 dwords per row
      int k2 = idx & 255;
      *(unsigned*)&Al[m * KP + 2 * k2] = hin32[m * (H_DIM / 2) + k2];
    }
    const float* xt = input + ((size_t)t * BATCH + rowbase) * I_DIM;
    for (int idx = tid; idx < ROW_TILE * I_DIM; idx += WG_THREADS) {
      int m = idx >> 8;
      int i = idx & 255;
      Al[m * KP + H_DIM + i] = (bf16_t)xt[m * I_DIM + i];
    }
    // prefetch next timestep's input rows (global_prefetch_b8), 256B/thread
    if (t + 1 < T_STEPS) {
      const float* xn = input + ((size_t)(t + 1) * BATCH + rowbase) * I_DIM;
      __builtin_prefetch(xn + (size_t)tid * 64, 0, 1);
    }
    __syncthreads();

    // ---- D(16x16, per wave) = A(16x768) x W(768x16), 24 x wmma.bf16 ----
    // Software-pipelined: chunk k+1's ds_loads issue before chunk k's WMMA so
    // LDS latency overlaps matrix-core execution on the scan's critical path.
    const bf16_t* Arow = Al + (size_t)n * KP;                 // A: lane&15 = M row
    const bf16_t* Bcol = Wl + (size_t)(wv * 16 + n) * KP;     // B: lane&15 = N col
    v8f acc = {};
    // A 16-bit 16x32 layout: lane half 'hi' holds K runs [k0+8hi..] and [k0+16+8hi..]
    // B 16-bit 32x16 layout: same K order as A: [k0+8hi..] and [k0+16+8hi..]
    v8bf alo = *(const v8bf*)(Arow + hi * 8);
    v8bf ahi = *(const v8bf*)(Arow + 16 + hi * 8);
    v8bf blo = *(const v8bf*)(Bcol + hi * 8);
    v8bf bhi = *(const v8bf*)(Bcol + 16 + hi * 8);
#pragma unroll
    for (int kc = 0; kc < KCHUNKS; ++kc) {
      v16bf a = __builtin_shufflevector(alo, ahi, 0,1,2,3,4,5,6,7,8,9,10,11,12,13,14,15);
      v16bf b = __builtin_shufflevector(blo, bhi, 0,1,2,3,4,5,6,7,8,9,10,11,12,13,14,15);
      if (kc + 1 < KCHUNKS) {           // issue next chunk's loads before this WMMA
        const int k1 = (kc + 1) * 32;
        alo = *(const v8bf*)(Arow + k1 + hi * 8);
        ahi = *(const v8bf*)(Arow + k1 + 16 + hi * 8);
        blo = *(const v8bf*)(Bcol + k1 + hi * 8);
        bhi = *(const v8bf*)(Bcol + k1 + 16 + hi * 8);
      }
      acc = __builtin_amdgcn_wmma_f32_16x16x32_bf16(
          /*neg_a=*/false, a, /*neg_b=*/false, b,
          /*c_mod=*/(short)0, acc, /*reuse_a=*/false, /*reuse_b=*/false);
    }

    // ---- epilogue: h_{t+1} = tanh(acc); D layout: VGPR j -> M = j + 8*hi ----
    bf16_t* hout = hbuf + ((size_t)rg * 2 + ((t + 1) & 1)) * (ROW_TILE * H_DIM);
    const int gcol = colbase + wv * 16 + n;
    float* outt = out + (size_t)t * (BATCH * H_DIM);
    float hv[8];
#pragma unroll
    for (int j = 0; j < 8; ++j) {
      const int m = j + hi * 8;
      hv[j] = fast_tanh(acc[j]);
      outt[(size_t)(rowbase + m) * H_DIM + gcol] = hv[j];      // hiddens[t] (f32)
      hout[m * H_DIM + gcol] = (bf16_t)hv[j];                  // state for t+1 (bf16)
    }
    if (t == T_STEPS - 1) {                                    // h_last tail
      float* lastp = out + (size_t)T_STEPS * BATCH * H_DIM;
#pragma unroll
      for (int j = 0; j < 8; ++j)
        lastp[(size_t)(rowbase + j + hi * 8) * H_DIM + gcol] = hv[j];
    }
    __threadfence();   // release our h_{t+1} slice
    __syncthreads();
    if (tid == 0)
      __hip_atomic_fetch_add(counters + rg, 1u, __ATOMIC_RELEASE, __HIP_MEMORY_SCOPE_AGENT);
  }
}

extern "C" void kernel_launch(void* const* d_in, const int* in_sizes, int n_in,
                              void* d_out, int out_size, void* d_ws, size_t ws_size,
                              hipStream_t stream) {
  const float* input = (const float*)d_in[0];   // [2048, 64, 256]
  const float* w_in  = (const float*)d_in[1];   // [256, 512]
  const float* w_rec = (const float*)d_in[2];   // [512, 512]
  float* out = (float*)d_out;                   // [2048*64*512] ++ [64*512]

  unsigned* counters = (unsigned*)d_ws;                    // 4 x u32
  bf16_t*   hbuf     = (bf16_t*)((char*)d_ws + 256);       // 4 x 2 x 16 x 512 bf16 = 128 KB

  hipLaunchKernelGGL(rnn_init_counters, dim3(1), dim3(64), 0, stream, counters);

  const size_t shmem = ((size_t)COLS_WG * KP + (size_t)ROW_TILE * KP) * sizeof(bf16_t); // ~218 KB < 320 KB
  hipLaunchKernelGGL(rnn_scan, dim3(16), dim3(WG_THREADS), shmem, stream,
                     input, w_in, w_rec, out, counters, hbuf);
}
